// LSTM_49357764165950
// MI455X (gfx1250) — matrix operand order repaired and checked by reading, not compiled, not run on hardware
//
#include <hip/hip_runtime.h>
#include <hip/hip_bf16.h>

// LSTM scan on MI455X (gfx1250), bf16 WMMA with fp32 accumulate.
// B=1024, T=256 cell evals (tag + 255), E=128, H=256, 4H=1024.
//
// Per-step: gates = x@W_ih^T + h@W_hh^T + b  ->  fused LSTM cell update.
// - Weights: bf16 in L2 (768KB), streamed per step. A laundered *integer*
//   offset keeps the loads inside the t-loop (no LICM spill) while keeping
//   the global address space visible -> global_load_b128 (LOADcnt only,
//   no flat/DScnt coupling).
// - h: bf16 tile in LDS per workgroup (next step's WMMA A-matrix).
// - c: lives in WMMA accumulator-layout registers for all 256 steps.
// - Each workgroup owns 16 batch rows for the whole sequence (64 WGs).
// - 16 waves/WG; wave w owns hidden tile w (16 units) x 4 gates = 4 C-tiles,
//   so i/f/g/o land in matching lane/VGPR slots -> register-only cell update.
// - Accumulation chains start from the inline-constant 0 C-operand; bias is
//   folded into the activation epilogue (kills the per-step splat copies).

#define B_SZ    1024
#define T_STEPS 256
#define E_DIM   128
#define H_DIM   256
#define G_DIM   1024   // 4*H
#define SCALE_F 4.0f

typedef __attribute__((ext_vector_type(16))) __bf16 v16bf;
typedef __attribute__((ext_vector_type(8)))  __bf16 v8bf;
typedef __attribute__((ext_vector_type(8)))  float  v8f;

__device__ __forceinline__ unsigned short f2bf(float f) {
    union { float f; unsigned int u; } v; v.f = f;
    unsigned int u = v.u;
    return (unsigned short)((u + 0x7FFFu + ((u >> 16) & 1u)) >> 16);
}

__device__ __forceinline__ v16bf pack16(v8bf lo, v8bf hi) {
    v16bf r;
#pragma unroll
    for (int i = 0; i < 8; ++i) { r[i] = lo[i]; r[i + 8] = hi[i]; }
    return r;
}

// ---------------------------------------------------------------------------
// Prep: fp32 -> bf16 weights (row-major: K contiguous = WMMA B-layout ready),
// fused bias = b_ih + b_hh.
// ---------------------------------------------------------------------------
__global__ void lstm_prep(const float* __restrict__ W_ih,
                          const float* __restrict__ W_hh,
                          const float* __restrict__ b_ih,
                          const float* __restrict__ b_hh,
                          unsigned short* __restrict__ wih_bf,
                          unsigned short* __restrict__ whh_bf,
                          float* __restrict__ bias) {
    int i = blockIdx.x * blockDim.x + threadIdx.x;
    if (i < G_DIM * E_DIM) wih_bf[i] = f2bf(W_ih[i]);
    if (i < G_DIM * H_DIM) whh_bf[i] = f2bf(W_hh[i]);
    if (i < G_DIM)         bias[i]   = b_ih[i] + b_hh[i];
}

// ---------------------------------------------------------------------------
// Persistent recurrent kernel: 64 blocks x 512 threads (16 waves).
// ---------------------------------------------------------------------------
__global__ void __launch_bounds__(512)
lstm_recur(const float* __restrict__ observed,
           const float* __restrict__ W_emb,
           const float* __restrict__ b_emb,
           const unsigned short* __restrict__ wih_bf0,
           const unsigned short* __restrict__ whh_bf0,
           const float* __restrict__ bias,
           float* __restrict__ out) {
    __shared__ __align__(16) unsigned short xs[16 * E_DIM];  // x_t tile bf16, 4KB
    __shared__ __align__(16) unsigned short hs[16 * H_DIM];  // h tile bf16, 8KB

    const int tid  = threadIdx.x;
    const int lane = tid & 31;
    const int wave = tid >> 5;          // 0..15 == hidden tile index
    const int lm   = lane & 15;         // tile column / row-low
    const int lg   = lane >> 4;         // lane half (A: K-group, C: M-high)
    const int b0   = blockIdx.x * 16;   // first batch row of this workgroup

    // h0 = 0
    for (int i = tid; i < 16 * H_DIM; i += 512) hs[i] = 0;

    // c state in registers, WMMA C-layout (m = r + 8*lg, n = wave*16 + lm)
    v8f creg;
#pragma unroll
    for (int r = 0; r < 8; ++r) creg[r] = 0.0f;

    // per-lane gate-row indices + fused bias (loop-invariant, tiny)
    int   nrow[4];
    float bv[4];
#pragma unroll
    for (int g = 0; g < 4; ++g) {
        nrow[g] = g * H_DIM + wave * 16 + lm;   // row of W_ih / W_hh
        bv[g]   = bias[nrow[g]];
    }

    const v8f zero8 = {};   // inline-constant 0 C-operand for chain starts

    for (int t = 0; t < T_STEPS; ++t) {
        // Launder a zero *offset* (not the pointer): addresses become loop-
        // variant (defeats LICM/spill) but the base stays a kernarg global
        // pointer, so loads lower to global_load_b128 (LOADcnt only).
        unsigned int woff = 0;
        asm volatile("" : "+s"(woff));
        const unsigned short* wih_bf = wih_bf0 + woff;
        const unsigned short* whh_bf = whh_bf0 + woff;

        // -------- stage x_t into LDS (fused velocity -> relu embedding) ----
        // 16 rows x 128 cols; 32 threads per row, 4 elems per thread.
        {
            int row = tid >> 5;
            int e0  = (tid & 31) * 4;
            int b   = b0 + row;
            float o0 = 0.f, o1 = 0.f;
            if (t > 0) {
                size_t base = (size_t)(b * T_STEPS + t) * 2;
                o0 = observed[base + 0] - observed[base - 2];
                o1 = observed[base + 1] - observed[base - 1];
            }
            if (t + 1 < T_STEPS)
                __builtin_prefetch(&observed[(size_t)(b * T_STEPS + t + 1) * 2], 0, 1);
#pragma unroll
            for (int j = 0; j < 4; ++j) {
                int e = e0 + j;
                float val;
                if (t == 0) {
                    val = (e == E_DIM - 2) ? 1.0f : 0.0f;          // tag token
                } else if (e < E_DIM - 2) {
                    float v = SCALE_F * o0 * W_emb[e]
                            + SCALE_F * o1 * W_emb[(E_DIM - 2) + e]
                            + b_emb[e];
                    val = v > 0.f ? v : 0.f;                        // relu
                } else {
                    val = 0.f;                                      // zero pad
                }
                xs[row * E_DIM + e] = f2bf(val);
            }
        }
        __syncthreads();

        // -------- gates = x @ W_ih^T + h @ W_hh^T (WMMA), bias folded later
        v8f acc[4];

        // x-projection: K = 128 -> 4 k-tiles of 32
#pragma unroll
        for (int kt = 0; kt < 4; ++kt) {
            v8bf a0 = *(const v8bf*)&xs[lm * E_DIM + kt * 32 + lg * 8];
            v8bf a1 = *(const v8bf*)&xs[lm * E_DIM + kt * 32 + 16 + lg * 8];
            v16bf A = pack16(a0, a1);
#pragma unroll
            for (int g = 0; g < 4; ++g) {
                const unsigned short* p =
                    &wih_bf[(size_t)nrow[g] * E_DIM + kt * 32 + lg * 8];
                v16bf Bm = pack16(*(const v8bf*)p, *(const v8bf*)(p + 16));
                acc[g] = __builtin_amdgcn_wmma_f32_16x16x32_bf16(
                    false, A, false, Bm, (short)0,
                    (kt == 0) ? zero8 : acc[g],   // chain starts from inline 0
                    false, false);
            }
        }
        // h-projection: K = 256 -> 8 k-tiles of 32
#pragma unroll
        for (int kt = 0; kt < 8; ++kt) {
            v8bf a0 = *(const v8bf*)&hs[lm * H_DIM + kt * 32 + lg * 8];
            v8bf a1 = *(const v8bf*)&hs[lm * H_DIM + kt * 32 + 16 + lg * 8];
            v16bf A = pack16(a0, a1);
#pragma unroll
            for (int g = 0; g < 4; ++g) {
                const unsigned short* p =
                    &whh_bf[(size_t)nrow[g] * H_DIM + kt * 32 + lg * 8];
                v16bf Bm = pack16(*(const v8bf*)p, *(const v8bf*)(p + 16));
                acc[g] = __builtin_amdgcn_wmma_f32_16x16x32_bf16(
                    false, A, false, Bm, (short)0, acc[g], false, false);
            }
        }
        __syncthreads();   // all xs/hs reads for this step are done

        // -------- fused cell update (register-only, bias folded here) -----
        {
            int ncol = wave * 16 + lm;              // hidden index
#pragma unroll
            for (int r = 0; r < 8; ++r) {
                float si = 1.f / (1.f + __expf(-(acc[0][r] + bv[0])));
                float sf = 1.f / (1.f + __expf(-(acc[1][r] + bv[1])));
                float so = 1.f / (1.f + __expf(-(acc[3][r] + bv[3])));
                float cn = sf * creg[r] + si * tanhf(acc[2][r] + bv[2]);
                float hn = so * tanhf(cn);
                creg[r] = cn;
                int m = r + (lg << 3);              // batch row within tile
                hs[m * H_DIM + ncol] = f2bf(hn);    // feeds next step's A
                if (t == T_STEPS - 1) {
                    int b = b0 + m;
                    out[(size_t)b * H_DIM + ncol] = hn;                         // h
                    out[(size_t)B_SZ * H_DIM + (size_t)b * H_DIM + ncol] = cn;  // c
                }
            }
        }
        // No trailing barrier needed: next iteration's hs reads happen only
        // after the staging __syncthreads(), which orders these writes.
    }
}

// ---------------------------------------------------------------------------
extern "C" void kernel_launch(void* const* d_in, const int* in_sizes, int n_in,
                              void* d_out, int out_size, void* d_ws, size_t ws_size,
                              hipStream_t stream) {
    const float* observed = (const float*)d_in[0];
    // d_in[1] = pooled: unused by the reference
    const float* W_emb = (const float*)d_in[2];
    const float* b_emb = (const float*)d_in[3];
    const float* W_ih  = (const float*)d_in[4];
    const float* W_hh  = (const float*)d_in[5];
    const float* b_ih  = (const float*)d_in[6];
    const float* b_hh  = (const float*)d_in[7];

    // workspace: 256KB W_ih(bf16) | 512KB W_hh(bf16) | 4KB fused bias
    unsigned short* wih_bf = (unsigned short*)d_ws;
    unsigned short* whh_bf = (unsigned short*)((char*)d_ws + (256u << 10));
    float*          bias   = (float*)((char*)d_ws + (768u << 10));

    lstm_prep<<<(G_DIM * H_DIM + 255) / 256, 256, 0, stream>>>(
        W_ih, W_hh, b_ih, b_hh, wih_bf, whh_bf, bias);

    lstm_recur<<<B_SZ / 16, 512, 0, stream>>>(
        observed, W_emb, b_emb, wih_bf, whh_bf, bias, (float*)d_out);
}
